// STGCNBlock_29068338659936
// MI455X (gfx1250) — matrix operand order repaired and checked by reading, not compiled, not run on hardware
//
#include <hip/hip_runtime.h>
#include <hip/hip_bf16.h>

// ST-GCN block for MI455X (gfx1250, wave32, WMMA).
//
// Schedule (memory-bound, ~2.0 GB HBM traffic ~= 86us @ 23.3 TB/s):
//   K1: x[B,T,V,C] --(bf16 WMMA 16x16x32, f32 acc)--> sup --(adj mix + bias)-->
//       g[B,V,T,C] (f32) + deterministic per-WG BN1 partial {sum,sumsq}
//   R : tree-reduce partials -> fused scale/shift (BN1)
//   K2: relu(bn1(g)) staged bf16 with halo --(grouped conv as K=192 GEMM via
//       6x WMMA bf16)--> y[B,V,T,C] + deterministic BN2 partials
//   R : reduce -> scale/shift (BN2)
//   K3: out[B,T,V,C] = relu(bn2(y)) + x   (float4, coalesced)
//
// Precision: bf16 inputs to WMMA, f32 accumulate (fp32 WMMA 16x16x4 would make
// the GEMMs the bottleneck; bf16 16x16x32 keeps us memory-bound).
// Workspace requirement: 2*285MB (g,y) + 73MB (BN partials) + small ~= 645MB.
// No float atomics anywhere -> bit-deterministic across replays.

typedef __attribute__((ext_vector_type(4)))  __bf16 v4bf;
typedef __attribute__((ext_vector_type(8)))  __bf16 v8bf;
typedef __attribute__((ext_vector_type(16))) __bf16 v16bf;
typedef __attribute__((ext_vector_type(8)))  float  v8f;

constexpr int kB = 32, kT = 2048, kV = 17, kC = 64;
constexpr int kVC  = kV * kC;                 // 1088 BN channels
constexpr float kEps = 1e-5f;

constexpr int TT1   = 8;                      // (b,t) positions per WG, kernel 1
constexpr int R1    = TT1 * kV;               // 136 live rows
constexpr int R1P   = 144;                    // padded to 9 WMMA M-tiles of 16
constexpr int NWG1  = kB * (kT / TT1);        // 8192
constexpr int TT2   = 256;                    // timesteps per WG, kernel 2
constexpr int NTT2  = kT / TT2;               // 8
constexpr int NWG2  = kB * kV * NTT2;         // 4352
constexpr int NSLOT2 = kB * NTT2;             // 256 partial slots per channel

// ---------------------------------------------------------------------------
// WMMA bf16 fragment loaders (CDNA5 ISA 7.12.2 layouts).
// A (16x32, MxK): lane l: m=l&15, h=l>>4; elem j: K = (j>>3)*16 + h*8 + (j&7)
//   -> two contiguous 8-elem (16B) LDS reads at col0+h*8 and col0+16+h*8.
// B (32x16, KxN), weights stored transposed (K contiguous per column n):
//   lane l: n=l&15, h=l>>4; elem j: K = (j>>3)*16 + h*8 + (j&7) (same as A).
// ---------------------------------------------------------------------------
__device__ __forceinline__ v16bf load_a_frag(const __bf16* lds, int ld,
                                             int row0, int col0, int lane) {
  const int h = lane >> 4, m = lane & 15;
  const __bf16* p = lds + (row0 + m) * ld + col0 + h * 8;
  v8bf lo = *(const v8bf*)(p);
  v8bf hi = *(const v8bf*)(p + 16);
  v16bf r;
#pragma unroll
  for (int i = 0; i < 8; ++i) { r[i] = lo[i]; r[i + 8] = hi[i]; }
  return r;
}

__device__ __forceinline__ v16bf load_b_frag(const __bf16* ldsT, int ldk,
                                             int n0, int k0, int lane) {
  const int h = lane >> 4, n = lane & 15;
  const __bf16* p = ldsT + (n0 + n) * ldk + k0 + h * 8;
  v8bf lo = *(const v8bf*)(p);
  v8bf hi = *(const v8bf*)(p + 16);
  v16bf r;
#pragma unroll
  for (int i = 0; i < 8; ++i) { r[i] = lo[i]; r[i + 8] = hi[i]; }
  return r;
}

// ---------------------------------------------------------------------------
// K1: GCN (x @ W via WMMA, then 17x17 adjacency mix + bias) + BN1 partials.
// Writes g in [B, V, T, C] layout (temporal-conv friendly).
// ---------------------------------------------------------------------------
__global__ __launch_bounds__(256) void k1_gcn(
    const float* __restrict__ x, const float* __restrict__ adj,
    const float* __restrict__ gw, const float* __restrict__ gb,
    float* __restrict__ g, float* __restrict__ p1s, float* __restrict__ p1q) {
  __shared__ __bf16 As[R1P * kC];       // 18432 B  (x tile, bf16, zero-padded)
  __shared__ __bf16 Wt[kC * kC];        //  8192 B  (W transposed: Wt[d][c])
  __shared__ float  Sup[R1P * kC];      // 36864 B  (x @ W, f32; rows >=136 pad)
  __shared__ float  adj_s[kV * kV];     //  1156 B
  __shared__ float  gb_s[kC];

  const int tid = threadIdx.x;
  const int wg  = blockIdx.x;           // = b*(T/TT1) + t-tile
  const int b   = wg / (kT / TT1);
  const int t0  = (wg % (kT / TT1)) * TT1;

  // Stage transposed GCN weight: Wt[d][c] = gw[c*64 + d]  (16KB, L2-resident)
  for (int i = tid; i < kC * kC; i += 256) {
    int d = i >> 6, c = i & 63;
    Wt[i] = (__bf16)gw[c * kC + d];
  }
  for (int i = tid; i < kV * kV; i += 256) adj_s[i] = adj[i];
  if (tid < kC) gb_s[tid] = gb[tid];

  // Stage x tile: rows (bt,v) flattened, contiguous TT1*1088 floats.
  const float4* xp4 = (const float4*)(x + ((size_t)b * kT + t0) * (size_t)kVC);
  for (int i = tid; i < R1 * kC / 4; i += 256) {
    float4 t4 = xp4[i];
    v4bf pk; pk[0] = (__bf16)t4.x; pk[1] = (__bf16)t4.y;
             pk[2] = (__bf16)t4.z; pk[3] = (__bf16)t4.w;
    *(v4bf*)&As[i * 4] = pk;
  }
  for (int i = R1 * kC + tid; i < R1P * kC; i += 256) As[i] = (__bf16)0.0f;
  __syncthreads();

  const int lane = tid & 31, wave = tid >> 5;
  // 9 M-tiles x 4 N-tiles = 36 wave-tiles over 8 waves (nb fixed per wave,
  // so both B fragments stay hoisted in registers). K=64 -> 2 wmma.
  for (int tile = wave; tile < 36; tile += 8) {
    const int mb = tile >> 2, nb = tile & 3;
    v8f acc = {};
#pragma unroll
    for (int ks = 0; ks < 2; ++ks) {
      v16bf a = load_a_frag(As, kC, mb * 16, ks * 32, lane);
      v16bf w = load_b_frag(Wt, kC, nb * 16, ks * 32, lane);
      acc = __builtin_amdgcn_wmma_f32_16x16x32_bf16(
          false, a, false, w, (short)0, acc, false, false);
    }
    // C/D layout: VGPR r, lane l -> M = r + 8*(l>>4), N = l&15.
    // Sup is padded to R1P rows -> unconditional stores (no exec-mask churn).
    const int m0 = mb * 16 + ((lane >> 4) << 3);
    const int n  = nb * 16 + (lane & 15);
#pragma unroll
    for (int r = 0; r < 8; ++r) Sup[(m0 + r) * kC + n] = acc[r];
  }
  __syncthreads();

  // Adjacency mix + bias; write g; deterministic BN1 partials per channel.
  for (int ch = tid; ch < kVC; ch += 256) {
    const int u = ch >> 6, d = ch & 63;
    float s1 = 0.f, s2 = 0.f;
    for (int bt = 0; bt < TT1; ++bt) {
      float acc = gb_s[d];
#pragma unroll
      for (int vv = 0; vv < kV; ++vv)
        acc = fmaf(adj_s[u * kV + vv], Sup[(bt * kV + vv) * kC + d], acc);
      g[(((size_t)b * kV + u) * kT + (t0 + bt)) * kC + d] = acc;
      s1 += acc; s2 = fmaf(acc, acc, s2);
    }
    p1s[(size_t)wg * kVC + ch] = s1;   // every (slot,ch) written each launch
    p1q[(size_t)wg * kVC + ch] = s2;
  }
}

// ---------------------------------------------------------------------------
// Deterministic BN partial reduction -> fused scale/shift.
// part layout: [nslots][1088]; fixed per-thread order + fixed tree.
// ---------------------------------------------------------------------------
__global__ __launch_bounds__(256) void k_reduce_bn(
    const float* __restrict__ ps, const float* __restrict__ pq, int nslots,
    const float* __restrict__ gamma, const float* __restrict__ beta,
    float* __restrict__ scale, float* __restrict__ shift, float inv_n) {
  __shared__ float rs[256], rq[256];
  const int ch = blockIdx.x, tid = threadIdx.x;
  float s = 0.f, q = 0.f;
  for (int s0 = tid; s0 < nslots; s0 += 256) {
    s += ps[(size_t)s0 * kVC + ch];
    q += pq[(size_t)s0 * kVC + ch];
  }
  rs[tid] = s; rq[tid] = q;
  __syncthreads();
  for (int off = 128; off > 0; off >>= 1) {
    if (tid < off) { rs[tid] += rs[tid + off]; rq[tid] += rq[tid + off]; }
    __syncthreads();
  }
  if (tid == 0) {
    float m   = rs[0] * inv_n;
    float var = rq[0] * inv_n - m * m;          // biased variance
    float sc  = gamma[ch] * rsqrtf(var + kEps);
    scale[ch] = sc;
    shift[ch] = beta[ch] - m * sc;
  }
}

// ---------------------------------------------------------------------------
// K2: relu(bn1(g)) -> grouped temporal conv (k=3) as K=192 GEMM via WMMA.
// One WG per (b, v, 256-timestep tile). Writes y[B,V,T,C] + BN2 partials.
// ---------------------------------------------------------------------------
__global__ __launch_bounds__(256) void k2_conv(
    const float* __restrict__ g, const float* __restrict__ cw,
    const float* __restrict__ cb, const float* __restrict__ sc1,
    const float* __restrict__ sh1, float* __restrict__ y,
    float* __restrict__ p2s, float* __restrict__ p2q) {
  __shared__ __bf16 Hs[(TT2 + 2) * kC];   // 33024 B, row r <-> t = t0-1+r
  __shared__ __bf16 W2[kC * 192];         // 24576 B, W2[o][tap*64+k]
  __shared__ float  sc_s[kC], shf_s[kC], cb_s[kC];
  __shared__ float  red_s[kC], red_q[kC];

  const int tid = threadIdx.x;
  const int wg  = blockIdx.x;
  const int tt  = wg % NTT2;
  const int bv  = wg / NTT2;              // b*17 + v
  const int v   = bv % kV;
  const int b   = bv / kV;
  const int t0  = tt * TT2;

  if (tid < kC) {
    sc_s[tid]  = sc1[v * kC + tid];
    shf_s[tid] = sh1[v * kC + tid];
    cb_s[tid]  = cb[v * kC + tid];
  }
  __syncthreads();

  // Stage activations with halo, fusing BN1 + ReLU + f32->bf16.
  const float* gp = g + (size_t)bv * kT * kC;
  for (int i = tid; i < (TT2 + 2) * (kC / 4); i += 256) {
    const int r  = i >> 4;
    const int c4 = (i & 15) * 4;
    const int t  = t0 - 1 + r;
    float4 val = make_float4(0.f, 0.f, 0.f, 0.f);
    if ((unsigned)t < (unsigned)kT) val = *(const float4*)(gp + (size_t)t * kC + c4);
    v4bf pk;
    pk[0] = (__bf16)fmaxf(fmaf(val.x, sc_s[c4 + 0], shf_s[c4 + 0]), 0.f);
    pk[1] = (__bf16)fmaxf(fmaf(val.y, sc_s[c4 + 1], shf_s[c4 + 1]), 0.f);
    pk[2] = (__bf16)fmaxf(fmaf(val.z, sc_s[c4 + 2], shf_s[c4 + 2]), 0.f);
    pk[3] = (__bf16)fmaxf(fmaf(val.w, sc_s[c4 + 3], shf_s[c4 + 3]), 0.f);
    *(v4bf*)&Hs[r * kC + c4] = pk;
  }
  // Stage per-joint weights transposed: W2[o][tap*64+k] = cw[((v*64+o)*64+k)*3+tap]
  const float* cwp = cw + (size_t)v * kC * kC * 3;
  for (int i = tid; i < kC * kC * 3; i += 256) {   // contiguous global reads
    const int tap = i % 3, ok = i / 3;
    const int k = ok & 63, o = ok >> 6;
    W2[o * 192 + tap * kC + k] = (__bf16)cwp[i];
  }
  __syncthreads();

  const int lane = tid & 31, wave = tid >> 5;
  const int nb = wave & 3;                 // wave's fixed N-tile
  const int n  = nb * 16 + (lane & 15);
  const float biasv = cb_s[n];
  float s1 = 0.f, s2 = 0.f;
  // 16 M-tiles x 4 N-tiles / 8 waves -> 8 tiles per wave, same columns.
  for (int i = 0; i < 8; ++i) {
    const int mb = (wave >> 2) + 2 * i;
    v8f acc = {};
#pragma unroll
    for (int ks = 0; ks < 6; ++ks) {       // K = 192 = 6 * 32
      const int tap = ks >> 1;
      const int c0  = (ks & 1) * 32;
      v16bf a = load_a_frag(Hs, kC, mb * 16 + tap, c0, lane);
      v16bf w = load_b_frag(W2, 192, nb * 16, ks * 32, lane);
      acc = __builtin_amdgcn_wmma_f32_16x16x32_bf16(
          false, a, false, w, (short)0, acc, false, false);
    }
    const int m0 = mb * 16 + ((lane >> 4) << 3);
    float* yp = y + ((size_t)bv * kT + t0 + m0) * kC + n;
#pragma unroll
    for (int r = 0; r < 8; ++r) {
      const float o = acc[r] + biasv;
      yp[(size_t)r * kC] = o;
      s1 += o; s2 = fmaf(o, o, s2);
    }
  }
  // Column partials: lanes l and l+16 hold the same N -> xor-16 combine.
  s1 += __shfl_xor(s1, 16, 32);
  s2 += __shfl_xor(s2, 16, 32);
  __syncthreads();
  if (wave < 4 && lane < 16) { red_s[n] = s1; red_q[n] = s2; }
  __syncthreads();
  if (wave >= 4 && lane < 16) { red_s[n] += s1; red_q[n] += s2; }
  __syncthreads();
  if (tid < kC) {   // slot = b*NTT2 + tt; every (slot, v-channel) written once
    const size_t slot = (size_t)(b * NTT2 + tt);
    p2s[slot * kVC + v * kC + tid] = red_s[tid];
    p2q[slot * kVC + v * kC + tid] = red_q[tid];
  }
}

// ---------------------------------------------------------------------------
// K3: out[b,t,v,c] = relu(bn2(y)) + x   (float4 elementwise)
// ---------------------------------------------------------------------------
__global__ __launch_bounds__(256) void k3_out(
    const float* __restrict__ y, const float* __restrict__ x,
    const float* __restrict__ sc2, const float* __restrict__ sh2,
    float* __restrict__ out) {
  const size_t q = (size_t)blockIdx.x * 256 + threadIdx.x;
  const size_t flat = q * 4;
  if (flat >= (size_t)kB * kT * kVC) return;
  const int c = (int)(flat & 63);
  const size_t rr = flat >> 6;           // (b*T + t)*V + v
  const int v = (int)(rr % kV);
  const size_t bt = rr / kV;             // b*T + t
  const int t = (int)(bt % kT);
  const int b = (int)(bt / kT);
  const float4 yv = *(const float4*)(y + (((size_t)b * kV + v) * kT + t) * kC + c);
  const float4 xv = *(const float4*)(x + flat);
  const int ch = v * kC + c;
  float4 o;
  o.x = fmaxf(fmaf(yv.x, sc2[ch + 0], sh2[ch + 0]), 0.f) + xv.x;
  o.y = fmaxf(fmaf(yv.y, sc2[ch + 1], sh2[ch + 1]), 0.f) + xv.y;
  o.z = fmaxf(fmaf(yv.z, sc2[ch + 2], sh2[ch + 2]), 0.f) + xv.z;
  o.w = fmaxf(fmaf(yv.w, sc2[ch + 3], sh2[ch + 3]), 0.f) + xv.w;
  *(float4*)(out + flat) = o;
}

// ---------------------------------------------------------------------------
extern "C" void kernel_launch(void* const* d_in, const int* in_sizes, int n_in,
                              void* d_out, int out_size, void* d_ws, size_t ws_size,
                              hipStream_t stream) {
  const float* x    = (const float*)d_in[0];
  const float* adj  = (const float*)d_in[1];
  const float* gw   = (const float*)d_in[2];
  const float* gb   = (const float*)d_in[3];
  const float* bn1g = (const float*)d_in[4];
  const float* bn1b = (const float*)d_in[5];
  const float* cw   = (const float*)d_in[6];
  const float* cb   = (const float*)d_in[7];
  const float* bn2g = (const float*)d_in[8];
  const float* bn2b = (const float*)d_in[9];
  float* out = (float*)d_out;

  const size_t nElem = (size_t)kB * kT * kVC;   // 71,303,168 floats
  float* ws  = (float*)d_ws;
  float* g   = ws;                 ws += nElem;
  float* y   = ws;                 ws += nElem;
  float* p1s = ws;                 ws += (size_t)NWG1 * kVC;
  float* p1q = ws;                 ws += (size_t)NWG1 * kVC;
  float* p2s = ws;                 ws += (size_t)NSLOT2 * kVC;
  float* p2q = ws;                 ws += (size_t)NSLOT2 * kVC;
  float* sc1 = ws;                 ws += kVC;
  float* sh1 = ws;                 ws += kVC;
  float* sc2 = ws;                 ws += kVC;
  float* sh2 = ws;                 ws += kVC;
  // requires ws_size >= ~645 MB; every workspace word is rewritten each launch
  // (no stale-state dependence, no init memsets needed).

  const float inv_n = 1.0f / (float)((size_t)kB * kT);

  k1_gcn<<<NWG1, 256, 0, stream>>>(x, adj, gw, gb, g, p1s, p1q);
  k_reduce_bn<<<kVC, 256, 0, stream>>>(p1s, p1q, NWG1, bn1g, bn1b, sc1, sh1, inv_n);
  k2_conv<<<NWG2, 256, 0, stream>>>(g, cw, cb, sc1, sh1, y, p2s, p2q);
  k_reduce_bn<<<kVC, 256, 0, stream>>>(p2s, p2q, NSLOT2, bn2g, bn2b, sc2, sh2, inv_n);
  const size_t nQuads = nElem / 4;
  k3_out<<<(int)((nQuads + 255) / 256), 256, 0, stream>>>(y, x, sc2, sh2, out);
}
